// JTNNEncoder_14946486190827
// MI455X (gfx1250) — compile-verified, matrix operand order checked by reading, not run on hardware
//
#include <hip/hip_runtime.h>
#include <hip/hip_bf16.h>

#define HID 256
#define DEPTH 10
#define NTREE 512

typedef __attribute__((ext_vector_type(16))) _Float16 v16h;
typedef __attribute__((ext_vector_type(8)))  float    v8f;

// (m,kk) -> half index inside a 512-half (32 lanes x 16 halves) WMMA fragment tile,
// per CDNA5 ISA 16-bit A-matrix layout (7.12.2). m in [0,16), kk in [0,32).
__device__ __forceinline__ int frag_off(int m, int kk) {
  int hi   = (kk >> 3) & 1;                       // which half-wave holds this K
  int lane = m + (hi << 4);
  int v    = ((kk >> 4) & 1) * 4 + ((kk & 6) >> 1); // VGPR index 0..7
  int j    = (v << 1) + (kk & 1);                   // half index within lane
  return lane * 16 + j;
}

__device__ __forceinline__ float sigm(float x) { return 1.0f / (1.0f + __expf(-x)); }

#define WMMA_F16(a, b, c) \
  __builtin_amdgcn_wmma_f32_16x16x32_f16(false, (a), false, (b), (short)0, (c), false, false)

// 2-stage rotated software pipeline: two fragment buffers are live at once, so
// copy-propagation cannot collapse the rotation; each WMMA consumes fragments
// loaded one full round earlier -> partial waitcnts, one round of latency hidden.
template <int KB>
__device__ __forceinline__ v8f gemm_tile(const _Float16* __restrict__ ab,
                                         const _Float16* __restrict__ bb, v8f acc) {
  static_assert((KB & 1) == 0, "KB must be even");
  v16h a0 = *(const v16h*)(ab);
  v16h b0 = *(const v16h*)(bb);
  v16h a1 = *(const v16h*)(ab + 512);
  v16h b1 = *(const v16h*)(bb + 512);
#pragma unroll
  for (int kb = 0; kb < KB - 2; kb += 2) {
    acc = WMMA_F16(a0, b0, acc);                // waits for loads from last round only
    a0  = *(const v16h*)(ab + (kb + 2) * 512);  // refill while wmma(a1,b1) runs
    b0  = *(const v16h*)(bb + (kb + 2) * 512);
    acc = WMMA_F16(a1, b1, acc);
    a1  = *(const v16h*)(ab + (kb + 3) * 512);
    b1  = *(const v16h*)(bb + (kb + 3) * 512);
  }
  acc = WMMA_F16(a0, b0, acc);
  acc = WMMA_F16(a1, b1, acc);
  return acc;
}

// ---- pack a KxN f32 row-major weight into WMMA B-fragment layout (f16) ----
__global__ void k_pack(const float* __restrict__ W, _Float16* __restrict__ P, int K, int N) {
  int idx = blockIdx.x * 256 + threadIdx.x;
  if (idx >= K * N) return;
  int k = idx / N, n = idx - k * N;
  int nt = n >> 4, nl = n & 15, kb = k >> 5, kk = k & 31;
  int KB = K >> 5;
  P[(size_t)(nt * KB + kb) * 512 + frag_off(nl, kk)] = (_Float16)W[idx];
}

// ---- x = emb[fnode[fmess]] as f16 ----
__global__ void k_prep_x(const int* __restrict__ fmess, const int* __restrict__ fnode,
                         const float* __restrict__ emb, _Float16* __restrict__ xh, int M) {
  int idx = blockIdx.x * 256 + threadIdx.x;
  int m = idx >> 8, c = idx & 255;
  if (m >= M) return;
  int tok = fnode[fmess[m]];
  xh[idx] = (_Float16)emb[tok * HID + c];
}

__global__ void k_zero(float4* __restrict__ p, int n4) {
  int i = blockIdx.x * 256 + threadIdx.x;
  if (i < n4) p[i] = make_float4(0.f, 0.f, 0.f, 0.f);
}

// ---- r1 = x @ Wr + bUr  (bUr folded in; used inside sigmoid every step) ----
template <bool EDGE>
__global__ __launch_bounds__(256) void k_r1(const _Float16* __restrict__ xh,
                                            const _Float16* __restrict__ WrP,
                                            const float* __restrict__ bUr,
                                            float* __restrict__ r1, int M, int m0base) {
  extern __shared__ char smem[];
  _Float16* aF = (_Float16*)smem;               // 2 rowtiles * 8 kblks * 512 halves
  int tid = threadIdx.x;
  int m0 = m0base + blockIdx.x * 32;
  for (int idx = tid; idx < 32 * 256; idx += 256) {
    int m = idx >> 8, c = idx & 255;
    int mg = m0 + m;
    _Float16 v;
    if (EDGE) {
      v = (mg < M) ? xh[mg * 256 + c] : (_Float16)0.f;
    } else {
      v = xh[mg * 256 + c];
    }
    aF[(m >> 4) * 4096 + (c >> 5) * 512 + frag_off(m & 15, c & 31)] = v;
  }
  __syncthreads();
  int w = tid >> 5, lane = tid & 31;
  int nl = lane & 15, hi8 = (lane >> 4) * 8;
  for (int t = 0; t < 4; ++t) {
    int tile = w * 4 + t, rt = tile >> 4, nt = tile & 15;
    v8f acc = {};
    acc = gemm_tile<8>(aF + rt * 4096 + lane * 16, WrP + nt * 8 * 512 + lane * 16, acc);
    int c = nt * 16 + nl;
    float bc = bUr[c];
#pragma unroll
    for (int i = 0; i < 8; ++i) {
      int mg = m0 + rt * 16 + hi8 + i;
      if (!EDGE || mg < M) r1[mg * 256 + c] = acc[i] + bc;
    }
  }
}

// ---- one GRU step over all messages ----
template <bool EDGE>
__global__ __launch_bounds__(256) void k_gru(
    const _Float16* __restrict__ xh, const float* __restrict__ hin,
    const int* __restrict__ mess_graph, const float* __restrict__ r1,
    const _Float16* __restrict__ WzP, const float* __restrict__ bz,
    const _Float16* __restrict__ UrP,
    const _Float16* __restrict__ WhP, const float* __restrict__ bh,
    float* __restrict__ hout, int M, int m0base) {
  extern __shared__ char smem[];
  _Float16* hneiF = (_Float16*)smem;            // 8 tiles (d*2+rt) * 8 kblks * 512 = 32768 halves
  _Float16* aF    = hneiF + 8 * 4096;           // 2 rowtiles * 16 kblks * 512  = 16384 halves
  float*    sumh  = (float*)(aF + 2 * 8192);    // 32 x 256 f32
  int*      nidx  = (int*)(sumh + 8192);        // 32 x 4 neighbor indices
  int tid = threadIdx.x;
  int m0 = m0base + blockIdx.x * 32;

  if (tid < 128) {
    int mg = m0 + (tid >> 2);
    // out-of-range rows point at message 0, whose h is always 0
    nidx[tid] = (!EDGE || mg < M) ? mess_graph[mg * 4 + (tid & 3)] : 0;
  }
  __syncthreads();

  // stage x frags (kblks 0..7), gather h_nei frags, build sum_h (f32 + f16 into kblks 8..15)
  for (int idx = tid; idx < 32 * 256; idx += 256) {
    int m = idx >> 8, c = idx & 255;
    int rt = m >> 4, kb = c >> 5;
    int off = frag_off(m & 15, c & 31);
    int mg = m0 + m;
    _Float16 xv;
    if (EDGE) {
      xv = (mg < M) ? xh[mg * 256 + c] : (_Float16)0.f;
    } else {
      xv = xh[mg * 256 + c];
    }
    aF[rt * 8192 + kb * 512 + off] = xv;
    float s = 0.f;
#pragma unroll
    for (int d = 0; d < 4; ++d) {
      int mi = nidx[m * 4 + d];
      float v = hin[mi * 256 + c];
      s += v;
      hneiF[(d * 2 + rt) * 4096 + kb * 512 + off] = (_Float16)v;
    }
    sumh[m * 256 + c] = s;
    aF[rt * 8192 + (8 + kb) * 512 + off] = (_Float16)s;
  }
  __syncthreads();

  int w = tid >> 5, lane = tid & 31;
  int nl = lane & 15, hi8 = (lane >> 4) * 8;

  float zv[4][8];   // z gate, kept in registers across both GEMMs
  float sg[4][8];   // sum of r*h_nei

  for (int t = 0; t < 4; ++t) {
    int tile = w * 4 + t, rt = tile >> 4, nt = tile & 15;
    int c = nt * 16 + nl;

    // z = sigmoid([x|sum_h] @ Wz + bz), K = 512
    v8f acc = {};
    acc = gemm_tile<16>(aF + rt * 8192 + lane * 16, WzP + nt * 16 * 512 + lane * 16, acc);
    float bzc = bz[c];
#pragma unroll
    for (int i = 0; i < 8; ++i) zv[t][i] = sigm(acc[i] + bzc);

    // r = sigmoid(r1 + h_nei@Ur); sum_gated = sum_d r*h_nei
    float r1v[8];
#pragma unroll
    for (int i = 0; i < 8; ++i) {
      int mg = m0 + rt * 16 + hi8 + i;
      if (EDGE) {
        r1v[i] = (mg < M) ? r1[mg * 256 + c] : 0.f;
      } else {
        r1v[i] = r1[mg * 256 + c];
      }
      sg[t][i] = 0.f;
    }
#pragma unroll
    for (int d = 0; d < 4; ++d) {
      v8f racc = {};
      racc = gemm_tile<8>(hneiF + (d * 2 + rt) * 4096 + lane * 16,
                          UrP + nt * 8 * 512 + lane * 16, racc);
      const _Float16* hft = hneiF + (d * 2 + rt) * 4096 + (c >> 5) * 512;
#pragma unroll
      for (int i = 0; i < 8; ++i) {
        float rr = sigm(racc[i] + r1v[i]);
        float hv = (float)hft[frag_off(hi8 + i, c & 31)];
        sg[t][i] += rr * hv;
      }
    }
  }
  __syncthreads();  // all z-GEMM reads of aF kblks 8..15 are done

  // overwrite aF kblks 8..15 with sum_gated (f16)
  for (int t = 0; t < 4; ++t) {
    int tile = w * 4 + t, rt = tile >> 4, nt = tile & 15;
    int c = nt * 16 + nl;
#pragma unroll
    for (int i = 0; i < 8; ++i)
      aF[rt * 8192 + (8 + (c >> 5)) * 512 + frag_off(hi8 + i, c & 31)] = (_Float16)sg[t][i];
  }
  __syncthreads();

  // pre_h = tanh([x|sum_gated] @ Wh + bh);  h = ((1-z)*sum_h + z*pre_h)*mask
  for (int t = 0; t < 4; ++t) {
    int tile = w * 4 + t, rt = tile >> 4, nt = tile & 15;
    int c = nt * 16 + nl;
    v8f acc = {};
    acc = gemm_tile<16>(aF + rt * 8192 + lane * 16, WhP + nt * 16 * 512 + lane * 16, acc);
    float bhc = bh[c];
#pragma unroll
    for (int i = 0; i < 8; ++i) {
      int ml = rt * 16 + hi8 + i;
      int mg = m0 + ml;
      float pre = tanhf(acc[i] + bhc);
      float z = zv[t][i];
      float hval = (1.f - z) * sumh[ml * 256 + c] + z * pre;
      hval = (mg == 0) ? 0.f : hval;            // mask row 0 (branch-free)
      if (!EDGE || mg < M) hout[mg * 256 + c] = hval;
    }
  }
}

// ---- tree_vecs = relu([emb[fnode[root]] | sum_d messages[node_graph[root,d]]] @ Wo + bo) ----
__global__ __launch_bounds__(256) void k_root(
    const int* __restrict__ roots, const int* __restrict__ fnode,
    const int* __restrict__ node_graph, const float* __restrict__ emb,
    const float* __restrict__ messages,
    const _Float16* __restrict__ WoP, const float* __restrict__ bo,
    float* __restrict__ out) {
  extern __shared__ char smem[];
  _Float16* aF = (_Float16*)smem;               // 2 * 16 kblks * 512 halves
  int tid = threadIdx.x;
  int m0 = blockIdx.x * 32;
  for (int idx = tid; idx < 32 * 256; idx += 256) {
    int m = idx >> 8, c = idx & 255;
    int rt = m >> 4, kb = c >> 5;
    int off = frag_off(m & 15, c & 31);
    int rn = roots[m0 + m];
    int tok = fnode[rn];
    aF[rt * 8192 + kb * 512 + off] = (_Float16)emb[tok * 256 + c];
    float s = 0.f;
#pragma unroll
    for (int d = 0; d < 4; ++d) {
      int ni = node_graph[rn * 4 + d];
      s += messages[ni * 256 + c];
    }
    aF[rt * 8192 + (8 + kb) * 512 + off] = (_Float16)s;
  }
  __syncthreads();
  int w = tid >> 5, lane = tid & 31;
  int nl = lane & 15, hi8 = (lane >> 4) * 8;
  for (int t = 0; t < 4; ++t) {
    int tile = w * 4 + t, rt = tile >> 4, nt = tile & 15;
    v8f acc = {};
    acc = gemm_tile<16>(aF + rt * 8192 + lane * 16, WoP + nt * 16 * 512 + lane * 16, acc);
    int c = nt * 16 + nl;
    float bc = bo[c];
#pragma unroll
    for (int i = 0; i < 8; ++i) {
      float v = acc[i] + bc;
      out[(m0 + rt * 16 + hi8 + i) * 256 + c] = v > 0.f ? v : 0.f;
    }
  }
}

extern "C" void kernel_launch(void* const* d_in, const int* in_sizes, int n_in,
                              void* d_out, int out_size, void* d_ws, size_t ws_size,
                              hipStream_t stream) {
  const int*   fnode      = (const int*)d_in[0];
  const int*   fmess      = (const int*)d_in[1];
  const int*   node_graph = (const int*)d_in[2];
  const int*   mess_graph = (const int*)d_in[3];
  const int*   roots      = (const int*)d_in[4];
  const float* emb        = (const float*)d_in[5];
  const float* Wz         = (const float*)d_in[6];
  const float* bz         = (const float*)d_in[7];
  const float* Wr         = (const float*)d_in[8];
  const float* Ur         = (const float*)d_in[9];
  const float* bUr        = (const float*)d_in[10];
  const float* Wh         = (const float*)d_in[11];
  const float* bh         = (const float*)d_in[12];
  const float* Wo         = (const float*)d_in[13];
  const float* bo         = (const float*)d_in[14];

  const int M = in_sizes[1];                    // 200001 messages
  const size_t MH = (size_t)M * HID;

  char* ws = (char*)d_ws;
  size_t off = 0;
  auto alloc = [&](size_t bytes) -> void* {
    void* p = ws + off;
    off = (off + bytes + 255) & ~(size_t)255;
    return p;
  };
  _Float16* xh  = (_Float16*)alloc(MH * 2);     // x (f16)
  float*    r1  = (float*)alloc(MH * 4);        // x@Wr + bUr
  float*    h0  = (float*)alloc(MH * 4);        // ping-pong buffer (other is d_out msg region)
  _Float16* WzP = (_Float16*)alloc(512 * 256 * 2);
  _Float16* WrP = (_Float16*)alloc(256 * 256 * 2);
  _Float16* UrP = (_Float16*)alloc(256 * 256 * 2);
  _Float16* WhP = (_Float16*)alloc(512 * 256 * 2);
  _Float16* WoP = (_Float16*)alloc(512 * 256 * 2);

  float* tree_out = (float*)d_out;
  float* mess_out = tree_out + NTREE * HID;

  // pack weights into WMMA B-fragment layout (one-time per launch, tiny)
  k_pack<<<512, 256, 0, stream>>>(Wz, WzP, 512, 256);
  k_pack<<<256, 256, 0, stream>>>(Wr, WrP, 256, 256);
  k_pack<<<256, 256, 0, stream>>>(Ur, UrP, 256, 256);
  k_pack<<<512, 256, 0, stream>>>(Wh, WhP, 512, 256);
  k_pack<<<512, 256, 0, stream>>>(Wo, WoP, 512, 256);

  // x = emb[fnode[fmess]]
  k_prep_x<<<(int)((MH + 255) / 256), 256, 0, stream>>>(fmess, fnode, emb, xh, M);

  // h starts at zero in the d_out messages region (parity puts step 9 back there)
  int n4 = (int)(MH / 4);
  k_zero<<<(n4 + 255) / 256, 256, 0, stream>>>((float4*)mess_out, n4);

  int gridM   = (M + 31) / 32;                  // 6251
  int gridInt = (M % 32 == 0) ? gridM : gridM - 1;  // interior (fully in-range) blocks
  int edgeM0  = gridInt * 32;

  if (gridInt > 0)
    k_r1<false><<<gridInt, 256, 16384, stream>>>(xh, WrP, bUr, r1, M, 0);
  if (gridInt < gridM)
    k_r1<true><<<1, 256, 16384, stream>>>(xh, WrP, bUr, r1, M, edgeM0);

  float* hin = mess_out;
  float* hother = h0;
  for (int s = 0; s < DEPTH; ++s) {
    float* hout = hother;                       // s even -> h0, s odd -> mess_out; s=9 -> mess_out
    if (gridInt > 0)
      k_gru<false><<<gridInt, 256, 131584, stream>>>(xh, hin, mess_graph, r1,
                                                     WzP, bz, UrP, WhP, bh, hout, M, 0);
    if (gridInt < gridM)
      k_gru<true><<<1, 256, 131584, stream>>>(xh, hin, mess_graph, r1,
                                              WzP, bz, UrP, WhP, bh, hout, M, edgeM0);
    hother = hin;
    hin = hout;
  }

  // only the 512 root rows of node_vecs are needed
  k_root<<<NTREE / 32, 256, 32768, stream>>>(roots, fnode, node_graph, emb, mess_out,
                                             WoP, bo, tree_out);
}
